// HyperbolicAttention_56341380989447
// MI455X (gfx1250) — compile-verified, matrix operand order checked by reading, not run on hardware
//
#include <hip/hip_runtime.h>
#include <hip/hip_bf16.h>
#include <stddef.h>

// ---------------------------------------------------------------------------
// Hyperbolic graph attention, fused for MI455X (gfx1250, wave32, WMMA).
//   N = 12288 nodes, D = 128 features.
// ---------------------------------------------------------------------------

#define Nn 12288
#define Dd 128
#define EPSF 1e-7f
#define MAXN 0.999f         // 1.0 - 1e-3
#define LRELU_SLOPE 0.2f
#define LOG2E 1.44269504088896340f
#define LN2 0.69314718055994531f

typedef __attribute__((ext_vector_type(16))) _Float16 v16h;
typedef __attribute__((ext_vector_type(8)))  _Float16 v8h;
typedef __attribute__((ext_vector_type(8)))  float    v8f;

// ------------------------------- Kernel 1 ----------------------------------
// Per row: x_tan = arctanh(||x||) * x / ||x||; wh1 = x_tan . a[:D];
// wh2 = x_tan . a[D:]; store x_tan transposed as f16 [D, N] for WMMA B tiles.
__global__ __launch_bounds__(256) void hyp_attn_rowprep(
    const float* __restrict__ x, const float* __restrict__ avec,
    float* __restrict__ wh1, float* __restrict__ wh2,
    _Float16* __restrict__ xT) {
  const int lane = threadIdx.x & 31;
  const int wave = threadIdx.x >> 5;
  const int row = blockIdx.x * 8 + wave;

  const float4 xv = *(const float4*)(x + (size_t)row * Dd + lane * 4);

  float ss = xv.x * xv.x + xv.y * xv.y + xv.z * xv.z + xv.w * xv.w;
  #pragma unroll
  for (int m = 1; m < 32; m <<= 1) ss += __shfl_xor(ss, m, 32);

  const float nrm = __builtin_amdgcn_sqrtf(ss);
  float nc = fminf(fmaxf(nrm, EPSF), 1.0f - EPSF);
  // arctanh(nc) = 0.5 * ln((1+nc)/(1-nc))
  const float at = 0.5f * LN2 * __builtin_amdgcn_logf((1.0f + nc) / (1.0f - nc));
  const float scl = at / fmaxf(nrm, EPSF);

  float t0 = xv.x * scl, t1 = xv.y * scl, t2 = xv.z * scl, t3 = xv.w * scl;

  const float4 a1 = *(const float4*)(avec + lane * 4);
  const float4 a2 = *(const float4*)(avec + Dd + lane * 4);
  float d1 = t0 * a1.x + t1 * a1.y + t2 * a1.z + t3 * a1.w;
  float d2 = t0 * a2.x + t1 * a2.y + t2 * a2.z + t3 * a2.w;
  #pragma unroll
  for (int m = 1; m < 32; m <<= 1) {
    d1 += __shfl_xor(d1, m, 32);
    d2 += __shfl_xor(d2, m, 32);
  }
  if (lane == 0) {
    wh1[row] = d1;
    wh2[row] = d2;
  }
  // Transposed f16 store (column `row` of xT[D][N]); small total (3 MB).
  const int d0 = lane * 4;
  xT[(size_t)(d0 + 0) * Nn + row] = (_Float16)t0;
  xT[(size_t)(d0 + 1) * Nn + row] = (_Float16)t1;
  xT[(size_t)(d0 + 2) * Nn + row] = (_Float16)t2;
  xT[(size_t)(d0 + 3) * Nn + row] = (_Float16)t3;
}

// ------------------------------- Kernel 2 ----------------------------------
// Global max of wh2 -> single float (per-row softmax shift bound).
__global__ __launch_bounds__(256) void hyp_attn_maxw2(
    const float* __restrict__ wh2, float* __restrict__ maxout) {
  __shared__ float smax[8];
  float m = -3.0e38f;
  for (int j = threadIdx.x; j < Nn; j += 256) m = fmaxf(m, wh2[j]);
  #pragma unroll
  for (int s = 1; s < 32; s <<= 1) m = fmaxf(m, __shfl_xor(m, s, 32));
  if ((threadIdx.x & 31) == 0) smax[threadIdx.x >> 5] = m;
  __syncthreads();
  if (threadIdx.x == 0) {
    float r = smax[0];
    #pragma unroll
    for (int w = 1; w < 8; ++w) r = fmaxf(r, smax[w]);
    maxout[0] = r;
  }
}

// ------------------------------- Kernel 3 ----------------------------------
// Fused masked softmax + (attention @ x_tan) + expmap0/proj epilogue.
// One wave per 16-row tile, 8 waves per block (128 rows / block, 96 blocks).
// p_ij = exp(leakyrelu(wh1_i + wh2_j) - bound_i) if adj_ij else 0 — computed
// directly in the v_wmma 16x16x32 f16 A-operand register layout.
__global__ __launch_bounds__(256) void hyp_attn_fused(
    const int* __restrict__ adj, const float* __restrict__ wh1,
    const float* __restrict__ maxw2, const _Float16* __restrict__ xT,
    float* __restrict__ out) {
  __shared__ float s_wh2[Nn];  // 48 KB of 320 KB WGP LDS

  // Stage wh2 into LDS once (coalesced), shared by all 8 waves.
  {
    const float* wh2g = wh1 + Nn;  // wh2 is laid out right after wh1 in ws
    for (int j = threadIdx.x; j < Nn; j += 256) s_wh2[j] = wh2g[j];
  }
  __syncthreads();

  const int lane = threadIdx.x & 31;
  const int wave = threadIdx.x >> 5;
  const int i0 = (blockIdx.x * 8 + wave) * 16;
  const int myrow = i0 + (lane & 15);
  const int off = (lane >> 4) << 3;  // K-run base: 0 (lanes 0-15) / 8 (16-31)

  const float wh1r = wh1[myrow];
  const float mx = maxw2[0];
  float b = wh1r + mx;
  const float bound = (b > 0.0f) ? b : LRELU_SLOPE * b;  // >= every row score

  const int* adjRow = adj + (size_t)myrow * Nn;

  const v8f vzero = {0.f, 0.f, 0.f, 0.f, 0.f, 0.f, 0.f, 0.f};
  v8f acc[8];
  #pragma unroll
  for (int t = 0; t < 8; ++t) acc[t] = vzero;
  float denom = 0.0f;

  // p-entry: mask -> leakyrelu -> shifted exp (v_exp_f32) -> f16
  #define PENT(ad, w2)                                                      \
    ({                                                                      \
      float _s = wh1r + (w2);                                               \
      _s = (_s > 0.0f) ? _s : LRELU_SLOPE * _s;                             \
      float _p = ((ad) > 0)                                                 \
                     ? __builtin_amdgcn_exp2f((_s - bound) * LOG2E)         \
                     : 0.0f;                                                \
      denom += _p;                                                          \
      (_Float16)_p;                                                         \
    })

  for (int j0 = 0; j0 < Nn; j0 += 32) {
    // adj tile, loaded directly in WMMA-A lane layout (4x b128 per lane)
    const int* ap = adjRow + j0 + off;
    int4 c0 = *(const int4*)(ap);
    int4 c1 = *(const int4*)(ap + 4);
    int4 c2 = *(const int4*)(ap + 16);
    int4 c3 = *(const int4*)(ap + 20);
    __builtin_prefetch(ap + 64, 0, 1);  // global_prefetch for next chunks

    // wh2 slice from LDS (ds_load_b128 x4)
    float4 w0 = *(const float4*)(&s_wh2[j0 + off]);
    float4 w1 = *(const float4*)(&s_wh2[j0 + off + 4]);
    float4 w2 = *(const float4*)(&s_wh2[j0 + off + 16]);
    float4 w3 = *(const float4*)(&s_wh2[j0 + off + 20]);

    v16h aop;
    aop[0] = PENT(c0.x, w0.x);  aop[1] = PENT(c0.y, w0.y);
    aop[2] = PENT(c0.z, w0.z);  aop[3] = PENT(c0.w, w0.w);
    aop[4] = PENT(c1.x, w1.x);  aop[5] = PENT(c1.y, w1.y);
    aop[6] = PENT(c1.z, w1.z);  aop[7] = PENT(c1.w, w1.w);
    aop[8] = PENT(c2.x, w2.x);  aop[9] = PENT(c2.y, w2.y);
    aop[10] = PENT(c2.z, w2.z); aop[11] = PENT(c2.w, w2.w);
    aop[12] = PENT(c3.x, w3.x); aop[13] = PENT(c3.y, w3.y);
    aop[14] = PENT(c3.z, w3.z); aop[15] = PENT(c3.w, w3.w);

    // 8 B tiles (D=128) from transposed f16 x_tan; contiguous b128 loads.
    #pragma unroll
    for (int t = 0; t < 8; ++t) {
      const _Float16* bp =
          xT + (size_t)(16 * t + (lane & 15)) * Nn + j0 + off;
      v8h b0 = *(const v8h*)(bp);
      v8h b1 = *(const v8h*)(bp + 16);
      v16h bop = __builtin_shufflevector(b0, b1, 0, 1, 2, 3, 4, 5, 6, 7, 8,
                                         9, 10, 11, 12, 13, 14, 15);
      acc[t] = __builtin_amdgcn_wmma_f32_16x16x32_f16(
          false, aop, false, bop, (short)0, acc[t], false, false);
    }
  }
  #undef PENT

  // ---- Epilogue: normalize rows, expmap0, proj, store -------------------
  // Each lane holds the denominator partial for row (lane&15) over its
  // K-half; merge halves with a cross-lane xor-16 (ds_bpermute).
  float dAll = denom + __shfl_xor(denom, 16, 32);

  #pragma unroll
  for (int r = 0; r < 8; ++r) {
    // C layout: v8f element r = row r (lanes 0-15) / row r+8 (lanes 16-31)
    float sA = __shfl(dAll, r, 32);
    float sB = __shfl(dAll, r + 8, 32);
    float inv = 1.0f / ((lane < 16) ? sA : sB);

    float vals[8];
    float ssq = 0.0f;
    #pragma unroll
    for (int t = 0; t < 8; ++t) {
      float v = acc[t][r] * inv;
      vals[t] = v;
      ssq += v * v;
    }
    // 16-lane reduction within each half (keeps the two rows separate)
    #pragma unroll
    for (int m = 1; m < 16; m <<= 1) ssq += __shfl_xor(ssq, m, 32);

    float nrm = __builtin_amdgcn_sqrtf(fmaxf(ssq, EPSF * EPSF));
    nrm = fmaxf(nrm, EPSF);
    // tanh(nrm) via exp2:  (1 - e^{-2x}) / (1 + e^{-2x})
    float e = __builtin_amdgcn_exp2f(-2.0f * nrm * LOG2E);
    float th = (1.0f - e) / (1.0f + e);
    // expmap0 scale tanh(n)/n ; proj clamps final norm (== tanh(n)) to MAXN
    float s = ((th > MAXN) ? MAXN : th) / nrm;

    const int rowm = i0 + r + ((lane >> 4) << 3);
    float* op = out + (size_t)rowm * Dd + (lane & 15);
    #pragma unroll
    for (int t = 0; t < 8; ++t) op[16 * t] = vals[t] * s;
  }
}

// ------------------------------- Launch ------------------------------------
extern "C" void kernel_launch(void* const* d_in, const int* in_sizes, int n_in,
                              void* d_out, int out_size, void* d_ws,
                              size_t ws_size, hipStream_t stream) {
  const float* x = (const float*)d_in[0];
  const int* adj = (const int*)d_in[1];
  const float* a = (const float*)d_in[2];
  float* out = (float*)d_out;

  // Workspace layout: wh1[N] f32 | wh2[N] f32 | maxw2 + pad | xT[D*N] f16
  float* wh1 = (float*)d_ws;
  float* wh2 = wh1 + Nn;
  float* maxp = wh2 + Nn;
  _Float16* xT = (_Float16*)(maxp + 64);  // 16B-aligned, 3 MB

  hipLaunchKernelGGL(hyp_attn_rowprep, dim3(Nn / 8), dim3(256), 0, stream,
                     x, a, wh1, wh2, xT);
  hipLaunchKernelGGL(hyp_attn_maxw2, dim3(1), dim3(256), 0, stream, wh2, maxp);
  hipLaunchKernelGGL(hyp_attn_fused, dim3(Nn / 128), dim3(256), 0, stream,
                     adj, wh1, maxp, xT, out);
}
